// b_multiheadattention_69733089017886
// MI455X (gfx1250) — matrix operand order repaired and checked by reading, not compiled, not run on hardware
//
#include <hip/hip_runtime.h>
#include <hip/hip_bf16.h>
#include <math.h>

// ---------------------------------------------------------------------------
// MI455X (gfx1250) cross-attention block.
// - All matmuls via v_wmma_f32_16x16x32_bf16 (wave32 WMMA, f32 accumulate).
// - GEMM tile staging via the Tensor Data Mover (tensor_load_to_lds, TENSORcnt)
//   with hardware LDS row padding (16dw interval + 4dw pad = 40-elem stride).
// - Attention staging via async global->LDS copies (ASYNCcnt).
// - Double-buffered LDS so copies overlap WMMA compute.
// ---------------------------------------------------------------------------

typedef unsigned short u16;
typedef __attribute__((ext_vector_type(16))) __bf16 v16bf;
typedef __attribute__((ext_vector_type(8)))  float  v8f;
typedef __attribute__((ext_vector_type(8)))  int    v8i;
typedef __attribute__((ext_vector_type(4)))  unsigned su4;
typedef __attribute__((ext_vector_type(8)))  unsigned su8;

#define H_HEADS 8
#define D_HEAD  1024
#define D_IMG   1024
#define D_SENT  768
#define HD      (H_HEADS * D_HEAD)   // 8192
#define BATCH   8
#define NQ      196
#define NK      128
#define MROWS   (BATCH * NQ)         // 1568
#define KROWS   (BATCH * NK)         // 1024

__device__ __forceinline__ u16 f2bf(float f) {
  unsigned u = __float_as_uint(f);
  u += 0x7FFFu + ((u >> 16) & 1u);   // round-to-nearest-even
  return (u16)(u >> 16);
}

// Low 32 bits of a generic LDS pointer = byte offset within the LDS aperture.
__device__ __forceinline__ unsigned lds_off(const void* p) {
  return (unsigned)(size_t)p;
}

// --- CDNA5 async global->LDS copies (ASYNCcnt) -----------------------------
__device__ __forceinline__ void async_g2l_b128(unsigned lds, const void* g) {
  asm volatile("global_load_async_to_lds_b128 %0, %1, off"
               :: "v"(lds), "v"((unsigned long long)(size_t)g) : "memory");
}
__device__ __forceinline__ void async_g2l_b32(unsigned lds, const void* g) {
  asm volatile("global_load_async_to_lds_b32 %0, %1, off"
               :: "v"(lds), "v"((unsigned long long)(size_t)g) : "memory");
}
__device__ __forceinline__ void wait_async0() {
  asm volatile("s_wait_asynccnt 0" ::: "memory");
}

// --- CDNA5 Tensor Data Mover: 2-D tile load, bf16 elements -----------------
// Pads LDS with 4 DWORDs after every 16 DWORDs stored: tile_dim0=32 elems
// (16 dw) per row -> effective LDS row stride = 40 elements (matches frags).
// OOB rows (tensor_dim1 exhausted) read as zero -> handles partial M tiles.
__device__ __forceinline__ void tdm_load_2d(unsigned lds_byte,
                                            const void* gaddr,
                                            unsigned tensor_d0,   // elems
                                            unsigned tensor_d1,   // rows
                                            unsigned tile_d0,     // elems
                                            unsigned tile_d1,     // rows
                                            unsigned d0_stride) { // elems
  unsigned long long ga = (unsigned long long)(size_t)gaddr;
  su4 g0;
  g0[0] = 1u;                                        // count=1, user mode
  g0[1] = lds_byte;                                  // lds_addr
  g0[2] = (unsigned)(ga & 0xFFFFFFFFu);              // global_addr[31:0]
  g0[3] = (unsigned)((ga >> 32) & 0x01FFFFFFu) | (2u << 30);  // type=2
  su8 g1;
  g1[0] = (1u << 16) | (1u << 20) | (3u << 22) | (3u << 25);
  //        data=2B     pad_en      interval=16dw   amount=4dw
  g1[1] = (tensor_d0 & 0xFFFFu) << 16;
  g1[2] = (tensor_d0 >> 16) | ((tensor_d1 & 0xFFFFu) << 16);
  g1[3] = (tensor_d1 >> 16) | (tile_d0 << 16);
  g1[4] = tile_d1;                                   // tile_dim2 = 0
  g1[5] = d0_stride;                                 // dim0_stride[31:0]
  g1[6] = 0;                                         // stride hi, dim1_stride lo
  g1[7] = 0;
  su4 z = (su4){0u, 0u, 0u, 0u};                     // groups 2/3: zeros (2-D)
  asm volatile("tensor_load_to_lds %0, %1, %2, %3"
               :: "s"(g0), "s"(g1), "s"(z), "s"(z) : "memory");
}
__device__ __forceinline__ void wait_tensor0() {
  __builtin_amdgcn_s_wait_tensorcnt(0);
}

// A-fragment (16x32 bf16, MxK): lane r=lane&15 is row; h=lane>>4 selects K
// half. VGPR0-3 = K[h*8..h*8+7], VGPR4-7 = K[16+h*8..16+h*8+7].
__device__ __forceinline__ v16bf frag_a(const u16* p, int stride, int r, int h) {
  v8i raw;
  ((int4*)&raw)[0] = *(const int4*)(p + r * stride + h * 8);
  ((int4*)&raw)[1] = *(const int4*)(p + r * stride + 16 + h * 8);
  return __builtin_bit_cast(v16bf, raw);
}

// B-fragment (32x16 bf16, KxN) from an LDS tile stored [N][K]:
// lane n=lane&15 is column; h=lane>>4 selects K half. VGPR0-3 = K[h*8..h*8+7], VGPR4-7 = K[16+h*8..16+h*8+7].
__device__ __forceinline__ v16bf frag_b(const u16* p, int stride, int n, int h) {
  v8i raw;
  ((int4*)&raw)[0] = *(const int4*)(p + n * stride + h * 8);
  ((int4*)&raw)[1] = *(const int4*)(p + n * stride + 16 + h * 8);
  return __builtin_bit_cast(v16bf, raw);
}

__device__ __forceinline__ v8f wmma_bf16(v16bf a, v16bf b, v8f c) {
  return __builtin_amdgcn_wmma_f32_16x16x32_bf16(
      false, a, false, b, (short)0, c, false, false);
}

// ---------------------------------------------------------------------------
// fp32 -> bf16 conversion kernels.
// ---------------------------------------------------------------------------
__global__ void __launch_bounds__(256)
cvt_transpose(const float* __restrict__ in, u16* __restrict__ out, int R, int C) {
  size_t id = (size_t)blockIdx.x * 256 + threadIdx.x;
  if (id < (size_t)R * C) {
    int r = (int)(id / C), c = (int)(id % C);
    out[(size_t)c * R + r] = f2bf(in[id]);
  }
}

__global__ void __launch_bounds__(256)
cvt_copy(const float* __restrict__ in, u16* __restrict__ out, size_t n) {
  size_t id = (size_t)blockIdx.x * 256 + threadIdx.x;
  if (id < n) out[id] = f2bf(in[id]);
}

// ---------------------------------------------------------------------------
// Tiled bf16 GEMM: C[M,N] = A[M,K] * Bt[N,K]^T, f32 accumulate.
// 128x128 block tile, 8 waves (4M x 2N), each wave 2x4 WMMA tiles.
// Staging: each wave TDM-loads its own 16-row slice of the A and B tiles
// (double-buffered; wait own TENSORcnt, then barrier -> all slices visible).
// ---------------------------------------------------------------------------
template <bool OUT_BF16>
__global__ void __launch_bounds__(256)
gemm_kernel(const u16* __restrict__ A, const u16* __restrict__ Bt,
            u16* __restrict__ Cbf, float* __restrict__ Cf,
            int M, int K, int N) {
  __shared__ __align__(16) u16 As[2][128 * 40];
  __shared__ __align__(16) u16 Bs[2][128 * 40];

  const int t    = threadIdx.x;
  const int lane = t & 31;
  const int w    = t >> 5;
  const int r15  = lane & 15;
  const int h    = lane >> 4;
  const int mblk = blockIdx.y * 128;
  const int nblk = blockIdx.x * 128;
  const int wm   = (w & 3) * 32;   // wave M offset in tile
  const int wn   = (w >> 2) * 64;  // wave N offset in tile

  // Per-wave TDM slice: 16 rows of the A tile and 16 rows of the B tile.
  const int arow = mblk + w * 16;
  const int brow = nblk + w * 16;
  int arem = M - arow; if (arem < 0) arem = 0;   // OOB rows load as zero
  const int brem = N - brow;

  v8f acc[2][4];
  #pragma unroll
  for (int i = 0; i < 2; ++i)
    #pragma unroll
    for (int j = 0; j < 4; ++j)
      acc[i][j] = (v8f){0.f, 0.f, 0.f, 0.f, 0.f, 0.f, 0.f, 0.f};

  auto stage = [&](int buf, int k0) {
    tdm_load_2d(lds_off(&As[buf][w * 16 * 40]), A + (size_t)arow * K + k0,
                (unsigned)(K - k0), (unsigned)arem, 32u, 16u, (unsigned)K);
    tdm_load_2d(lds_off(&Bs[buf][w * 16 * 40]), Bt + (size_t)brow * K + k0,
                (unsigned)(K - k0), (unsigned)brem, 32u, 16u, (unsigned)K);
  };

  const int nk = K >> 5;
  stage(0, 0);

  for (int it = 0; it < nk; ++it) {
    const int buf = it & 1;
    wait_tensor0();         // this wave's TDM slices are in LDS
    __syncthreads();        // all waves' slices visible; prev compute done
    if (it + 1 < nk) stage(buf ^ 1, (it + 1) << 5);

    v16bf af[2], bf[4];
    af[0] = frag_a(As[buf], 40, wm + r15, h);
    af[1] = frag_a(As[buf], 40, wm + 16 + r15, h);
    #pragma unroll
    for (int j = 0; j < 4; ++j) bf[j] = frag_b(Bs[buf], 40, wn + j * 16 + r15, h);
    #pragma unroll
    for (int i = 0; i < 2; ++i)
      #pragma unroll
      for (int j = 0; j < 4; ++j)
        acc[i][j] = wmma_bf16(af[i], bf[j], acc[i][j]);
  }

  // Epilogue: C layout -> row = h*8 + g, col = lane&15 (fully unrolled).
  #pragma unroll
  for (int i = 0; i < 2; ++i)
    #pragma unroll
    for (int j = 0; j < 4; ++j) {
      const v8f a = acc[i][j];
      #pragma unroll
      for (int g = 0; g < 8; ++g) {
        int row = mblk + wm + i * 16 + h * 8 + g;
        int col = nblk + wn + j * 16 + r15;
        if (row < M) {
          if constexpr (OUT_BF16) Cbf[(size_t)row * N + col] = f2bf(a[g]);
          else                    Cf[(size_t)row * N + col]  = a[g];
        }
      }
    }
}

// ---------------------------------------------------------------------------
// Attention: per (b, head, 16-row q tile): S = Q K^T / 32, masked softmax,
// ctx = P V. NK=128 fits on-chip -> single-pass softmax.
// QK^T staging is double-buffered async; V is transposed through VGPRs.
// ---------------------------------------------------------------------------
__global__ void __launch_bounds__(256)
attn_kernel(const u16* __restrict__ Qb, const u16* __restrict__ Kb,
            const u16* __restrict__ Vb, const int* __restrict__ mask,
            u16* __restrict__ ctx) {
  const int qt  = blockIdx.x;   // 0..12 (13 tiles of 16 covering 196)
  const int hid = blockIdx.y;   // head
  const int b   = blockIdx.z;   // batch

  __shared__ __align__(16) u16   Qs[2][16 * 40];
  __shared__ __align__(16) u16   Ks[2][128 * 40];
  __shared__            float    Sf[16 * 128];
  __shared__ __align__(16) u16   Pb[16 * 128];
  __shared__ __align__(16) u16   Vt[128 * 128];   // [d_local][key]

  const int t    = threadIdx.x;
  const int lane = t & 31;
  const int w    = t >> 5;       // wave 0..7 -> 16-key / 16-d slice
  const int r15  = lane & 15;
  const int h    = lane >> 4;

  // Staging coordinates.
  const int qrow_l = t >> 4;                    // Q strip row 0..15
  const int qcol   = (t & 15) * 2;              // 2 bf16 per thread (b32)
  int qsrc = qt * 16 + qrow_l; if (qsrc > NQ - 1) qsrc = NQ - 1;
  const u16* qbase = Qb + ((size_t)(b * NQ + qsrc)) * HD + hid * D_HEAD + qcol;
  int krow[2], kcol[2];
  #pragma unroll
  for (int i = 0; i < 2; ++i) {
    int cid = t + i * 256;
    krow[i] = cid >> 2;
    kcol[i] = (cid & 3) * 8;
  }

  auto stageQK = [&](int buf, int c) {
    async_g2l_b32(lds_off(&Qs[buf][qrow_l * 40 + qcol]), qbase + c);
    #pragma unroll
    for (int i = 0; i < 2; ++i)
      async_g2l_b128(lds_off(&Ks[buf][krow[i] * 40 + kcol[i]]),
                     Kb + ((size_t)(b * NK + krow[i])) * HD +
                         hid * D_HEAD + c + kcol[i]);
  };

  // ---- Phase 1: S = Q K^T over D=1024 ----
  v8f sacc = (v8f){0.f, 0.f, 0.f, 0.f, 0.f, 0.f, 0.f, 0.f};
  const int nc = D_HEAD >> 5;   // 32 K-steps
  stageQK(0, 0);
  for (int it = 0; it < nc; ++it) {
    const int buf = it & 1;
    wait_async0();
    __syncthreads();
    if (it + 1 < nc) stageQK(buf ^ 1, (it + 1) << 5);
    v16bf aq = frag_a(Qs[buf], 40, r15, h);
    v16bf bk = frag_b(Ks[buf], 40, w * 16 + r15, h);  // keys are the N dim
    sacc = wmma_bf16(aq, bk, sacc);
  }

  // ---- Phase 2: masked softmax over 128 keys ----
  const float scale = 0.03125f;  // 1/sqrt(1024)
  #pragma unroll
  for (int g = 0; g < 8; ++g) {
    int row = h * 8 + g;
    int key = w * 16 + r15;
    float v = sacc[g] * scale;
    if (mask[b * NK + key] == 0) v = -__builtin_inff();
    Sf[row * 128 + key] = v;
  }
  __syncthreads();
  if (t < 16) {
    float m = -__builtin_inff();
    for (int k = 0; k < 128; ++k) m = fmaxf(m, Sf[t * 128 + k]);
    float s = 0.f;
    for (int k = 0; k < 128; ++k) {
      float e = __expf(Sf[t * 128 + k] - m);
      s += e;
      Sf[t * 128 + k] = e;
    }
    float inv = 1.0f / s;
    for (int k = 0; k < 128; ++k) Pb[t * 128 + k] = f2bf(Sf[t * 128 + k] * inv);
  }
  __syncthreads();

  // ---- Phase 3: ctx = P V, d in chunks of 128 (one 16-wide slice/wave) ----
  for (int dchunk = 0; dchunk < D_HEAD; dchunk += 128) {
    #pragma unroll
    for (int i = 0; i < 8; ++i) {  // transpose V chunk into LDS [d][key]
      int id  = t + 256 * i;       // 0..2047
      int key = id >> 4, sub = id & 15, dcol = sub * 8;
      __align__(16) u16 tmp[8];
      *(int4*)tmp = *(const int4*)(Vb + ((size_t)(b * NK + key)) * HD +
                                   hid * D_HEAD + dchunk + dcol);
      #pragma unroll
      for (int e = 0; e < 8; ++e) Vt[(dcol + e) * 128 + key] = tmp[e];
    }
    __syncthreads();

    v8f cacc = (v8f){0.f, 0.f, 0.f, 0.f, 0.f, 0.f, 0.f, 0.f};
    #pragma unroll
    for (int kk = 0; kk < 128; kk += 32) {
      v16bf ap = frag_a(Pb + kk, 128, r15, h);
      v16bf bv = frag_b(Vt + kk, 128, w * 16 + r15, h);
      cacc = wmma_bf16(ap, bv, cacc);
    }
    #pragma unroll
    for (int g = 0; g < 8; ++g) {
      int qrow = qt * 16 + h * 8 + g;
      if (qrow < NQ) {
        int d = dchunk + w * 16 + r15;
        ctx[((size_t)(b * NQ + qrow)) * HD + hid * D_HEAD + d] = f2bf(cacc[g]);
      }
    }
    __syncthreads();
  }
}

// ---------------------------------------------------------------------------
// Residual add + LayerNorm over the last dim (1024), one block per row.
// ---------------------------------------------------------------------------
__global__ void __launch_bounds__(256)
ln_kernel(const float* __restrict__ proj, const float* __restrict__ img,
          const float* __restrict__ gamma, const float* __restrict__ beta,
          float* __restrict__ out) {
  const int row = blockIdx.x;
  const int t   = threadIdx.x;
  __shared__ float s1[256], s2[256];

  float lsum = 0.f, lsq = 0.f;
  for (int i = t; i < D_IMG; i += 256) {
    float x = proj[(size_t)row * D_IMG + i] + img[(size_t)row * D_IMG + i];
    lsum += x; lsq += x * x;
  }
  s1[t] = lsum; s2[t] = lsq;
  __syncthreads();
  for (int s = 128; s > 0; s >>= 1) {
    if (t < s) { s1[t] += s1[t + s]; s2[t] += s2[t + s]; }
    __syncthreads();
  }
  float mu  = s1[0] * (1.0f / D_IMG);
  float var = s2[0] * (1.0f / D_IMG) - mu * mu;
  float inv = rsqrtf(var + 1e-5f);
  for (int i = t; i < D_IMG; i += 256) {
    float x = proj[(size_t)row * D_IMG + i] + img[(size_t)row * D_IMG + i];
    out[(size_t)row * D_IMG + i] = (x - mu) * inv * gamma[i] + beta[i];
  }
}

// ---------------------------------------------------------------------------
// Host-side orchestration.
// ---------------------------------------------------------------------------
extern "C" void kernel_launch(void* const* d_in, const int* in_sizes, int n_in,
                              void* d_out, int out_size, void* d_ws, size_t ws_size,
                              hipStream_t stream) {
  (void)in_sizes; (void)n_in; (void)out_size; (void)ws_size;

  const float* img   = (const float*)d_in[0];   // [8,196,1024]
  const float* sent  = (const float*)d_in[1];   // [8,128,768]
  const int*   amask = (const int*)  d_in[2];   // [8,128]
  const float* Wq    = (const float*)d_in[3];   // [1024,8192]
  const float* Wk    = (const float*)d_in[4];   // [768,8192]
  const float* Wv    = (const float*)d_in[5];   // [768,8192]
  const float* Wo    = (const float*)d_in[6];   // [8192,1024]
  const float* gamma = (const float*)d_in[7];   // [1024]
  const float* beta  = (const float*)d_in[8];   // [1024]
  float* out = (float*)d_out;                   // [8,196,1024]

  // Workspace carve-up (256B aligned).
  char* base = (char*)d_ws;
  size_t off = 0;
  auto carve = [&](size_t bytes) -> char* {
    char* p = base + off;
    off = (off + bytes + 255) & ~(size_t)255;
    return p;
  };
  u16*  WqT   = (u16*)carve((size_t)HD * D_IMG  * 2);   // [8192][1024]
  u16*  WkT   = (u16*)carve((size_t)HD * D_SENT * 2);   // [8192][768]
  u16*  WvT   = (u16*)carve((size_t)HD * D_SENT * 2);   // [8192][768]
  u16*  WoT   = (u16*)carve((size_t)D_IMG * HD  * 2);   // [1024][8192]
  u16*  imgb  = (u16*)carve((size_t)MROWS * D_IMG  * 2);
  u16*  sentb = (u16*)carve((size_t)KROWS * D_SENT * 2);
  u16*  Qb    = (u16*)carve((size_t)MROWS * HD * 2);
  u16*  Kb    = (u16*)carve((size_t)KROWS * HD * 2);
  u16*  Vb    = (u16*)carve((size_t)KROWS * HD * 2);
  u16*  ctxb  = (u16*)carve((size_t)MROWS * HD * 2);
  float* proj = (float*)carve((size_t)MROWS * D_IMG * 4);

  auto blocks = [](size_t n) { return (unsigned)((n + 255) / 256); };

  // 1) fp32 -> bf16 conversions (weights transposed to [N][K]).
  cvt_transpose<<<blocks((size_t)D_IMG * HD), 256, 0, stream>>>(Wq, WqT, D_IMG, HD);
  cvt_transpose<<<blocks((size_t)D_SENT * HD), 256, 0, stream>>>(Wk, WkT, D_SENT, HD);
  cvt_transpose<<<blocks((size_t)D_SENT * HD), 256, 0, stream>>>(Wv, WvT, D_SENT, HD);
  cvt_transpose<<<blocks((size_t)HD * D_IMG), 256, 0, stream>>>(Wo, WoT, HD, D_IMG);
  cvt_copy<<<blocks((size_t)MROWS * D_IMG), 256, 0, stream>>>(img, imgb, (size_t)MROWS * D_IMG);
  cvt_copy<<<blocks((size_t)KROWS * D_SENT), 256, 0, stream>>>(sent, sentb, (size_t)KROWS * D_SENT);

  // 2) Projections (WMMA GEMMs, bf16 out, TDM-staged).
  gemm_kernel<true><<<dim3(HD / 128, (MROWS + 127) / 128), 256, 0, stream>>>(
      imgb, WqT, Qb, nullptr, MROWS, D_IMG, HD);
  gemm_kernel<true><<<dim3(HD / 128, KROWS / 128), 256, 0, stream>>>(
      sentb, WkT, Kb, nullptr, KROWS, D_SENT, HD);
  gemm_kernel<true><<<dim3(HD / 128, KROWS / 128), 256, 0, stream>>>(
      sentb, WvT, Vb, nullptr, KROWS, D_SENT, HD);

  // 3) Attention (WMMA for QK^T and PV; masked softmax on-chip).
  attn_kernel<<<dim3((NQ + 15) / 16, H_HEADS, BATCH), 256, 0, stream>>>(
      Qb, Kb, Vb, amask, ctxb);

  // 4) Output projection (fp32 out), then residual + LayerNorm.
  gemm_kernel<false><<<dim3(D_IMG / 128, (MROWS + 127) / 128), 256, 0, stream>>>(
      ctxb, WoT, nullptr, proj, MROWS, HD, D_IMG);
  ln_kernel<<<dim3(MROWS), 256, 0, stream>>>(proj, img, gamma, beta, out);
}
